// LSTM_78151224918778
// MI455X (gfx1250) — matrix operand order repaired and checked by reading, not compiled, not run on hardware
//
#include <hip/hip_runtime.h>

#define B_   32
#define T_   512
#define H_   1024
#define G4H_ 4096
#define NBLK 64

typedef __attribute__((ext_vector_type(16))) __bf16 v16bf;
typedef __attribute__((ext_vector_type(8)))  __bf16 v8bf;
typedef __attribute__((ext_vector_type(8)))  float  v8f;

__device__ __forceinline__ v8f wmma_bf16(v16bf a, v16bf b, v8f c) {
  return __builtin_amdgcn_wmma_f32_16x16x32_bf16(false, a, false, b, (short)0, c,
                                                 false, false);
}

// A fragment (16-bit A 16x32 ISA layout): lane holds K = {k0+half*8 .. +7} and
// {k0+16+half*8 .. +7} -> two contiguous 16-byte chunks.
__device__ __forceinline__ v16bf loadA(const __bf16* p /*row base + k0*/, int half) {
  v8bf a0 = *(const v8bf*)(p + half * 8);
  v8bf a1 = *(const v8bf*)(p + 16 + half * 8);
  return __builtin_shufflevector(a0, a1, 0, 1, 2, 3, 4, 5, 6, 7,
                                 8, 9, 10, 11, 12, 13, 14, 15);
}

// B fragment: lane holds K = {k0+half*8 .. +7} and {k0+16+half*8 .. +7}
// (same k order as A) -> two contiguous 16-byte chunks.
__device__ __forceinline__ v16bf loadB(const __bf16* p /*row base + k0*/, int half) {
  v8bf b0 = *(const v8bf*)(p + half * 8);
  v8bf b1 = *(const v8bf*)(p + 16 + half * 8);
  return __builtin_shufflevector(b0, b1, 0, 1, 2, 3, 4, 5, 6, 7,
                                 8, 9, 10, 11, 12, 13, 14, 15);
}

// ---------------------------------------------------------------------------
// Kernel: one-time fp32 -> (bf16 hi, bf16 lo) split; hi+lo = 4B/elem (same
// bytes as fp32, zero per-step conversion VALU).
// ---------------------------------------------------------------------------
__global__ void split_f32_bf16(const float* __restrict__ src,
                               __bf16* __restrict__ hi, __bf16* __restrict__ lo,
                               int nelem) {
  int idx = blockIdx.x * blockDim.x + threadIdx.x;
  int stride = gridDim.x * blockDim.x;
  for (int i = idx; i < nelem; i += stride) {
    float f = src[i];
    __bf16 h = (__bf16)f;
    hi[i] = h;
    lo[i] = (__bf16)(f - (float)h);
  }
}

// ---------------------------------------------------------------------------
// Kernel: zero h double-buffers (hi/lo bf16) and barrier words.
// ---------------------------------------------------------------------------
__global__ void lstm_init(unsigned* __restrict__ hzero, int nwords,
                          unsigned* __restrict__ bar) {
  int idx = blockIdx.x * blockDim.x + threadIdx.x;
  for (int i = idx; i < nwords; i += gridDim.x * blockDim.x) hzero[i] = 0u;
  if (idx == 0) { bar[0] = 0u; bar[1] = 0u; }
}

// ---------------------------------------------------------------------------
// Kernel: xg[t*32+b, col] = x[b,t,:] . w_ih[col,:] + b_ih[col] + b_hh[col]
// Split-bf16 3-WMMA (hi*hi + hi*lo + lo*hi ~ fp32 accuracy).
// ---------------------------------------------------------------------------
__global__ void __launch_bounds__(256)
lstm_xg_gemm(const __bf16* __restrict__ x_hi, const __bf16* __restrict__ x_lo,
             const __bf16* __restrict__ w_hi, const __bf16* __restrict__ w_lo,
             const float* __restrict__ b_ih, const float* __restrict__ b_hh,
             float* __restrict__ xg) {
  const int tid = threadIdx.x;
  const int wid = tid >> 5, lane = tid & 31, half = lane >> 4, n = lane & 15;
  const int rt = blockIdx.x;                 // row tile 0..1023 of [T*B]
  const int ct = blockIdx.y * 8 + wid;       // col tile 0..255
  const int t  = rt >> 1;
  const int b0 = (rt & 1) * 16;
  const int col = ct * 16 + n;

  const float bias = b_ih[col] + b_hh[col];
  v8f acc;
#pragma unroll
  for (int i = 0; i < 8; ++i) acc[i] = bias;

  const size_t arow = ((size_t)(b0 + n) * T_ + t) * H_;  // A row = b0 + (lane&15)
  const __bf16* ahp = x_hi + arow;
  const __bf16* alp = x_lo + arow;
  const __bf16* bhp = w_hi + (size_t)col * H_;           // B row = output column
  const __bf16* blp = w_lo + (size_t)col * H_;

#pragma unroll 4
  for (int kt = 0; kt < 32; ++kt) {
    const int k0 = kt * 32;
    v16bf ahi = loadA(ahp + k0, half);
    v16bf alo = loadA(alp + k0, half);
    v16bf bhi = loadB(bhp + k0, half);
    v16bf blo = loadB(blp + k0, half);
    acc = wmma_bf16(ahi, bhi, acc);
    acc = wmma_bf16(ahi, blo, acc);
    acc = wmma_bf16(alo, bhi, acc);
  }

  float* out = xg + (size_t)(rt * 16 + half * 8) * G4H_ + col;
#pragma unroll
  for (int i = 0; i < 8; ++i) out[(size_t)i * G4H_] = acc[i];
}

// ---------------------------------------------------------------------------
// Kernel: persistent scan. 64 blocks; block owns columns [jb, jb+16).
// Wave w: gate g = w>>1, row-half mh = w&1 -> one 16x16x1024 tile per step.
// h is produced directly as (hi,lo) bf16 by the cell update; c lives in LDS.
// One software grid barrier per step; h double-buffered in global (L2).
// ---------------------------------------------------------------------------
__global__ void __launch_bounds__(256)
lstm_scan(const float* __restrict__ xg,
          const __bf16* __restrict__ w_hi, const __bf16* __restrict__ w_lo,
          __bf16* __restrict__ h_hi,       // [2][32][1024]
          __bf16* __restrict__ h_lo,       // [2][32][1024]
          unsigned* __restrict__ bar,      // [0]=count, [1]=generation
          float* __restrict__ out) {
  __shared__ float gates[4][B_][16];
  __shared__ float cst[B_][16];

  const int tid = threadIdx.x;
  const int wid = tid >> 5, lane = tid & 31, half = lane >> 4, n = lane & 15;
  const int g = wid >> 1, mh = wid & 1;
  const int jb = blockIdx.x * 16;

  for (int idx = tid; idx < B_ * 16; idx += 256) cst[idx >> 4][idx & 15] = 0.f;
  __syncthreads();

  const __bf16* bhp = w_hi + (size_t)(g * H_ + jb + n) * H_;  // B row = column
  const __bf16* blp = w_lo + (size_t)(g * H_ + jb + n) * H_;

  for (int t = 0; t < T_; ++t) {
    const size_t src = (size_t)(t & 1) * (B_ * H_);

    // accumulator = precomputed input projection for this D tile
    const float* xrow = xg + (size_t)t * B_ * G4H_ +
                        (size_t)(mh * 16 + half * 8) * G4H_ + (g * H_ + jb + n);
    v8f acc;
#pragma unroll
    for (int i = 0; i < 8; ++i) acc[i] = xrow[(size_t)i * G4H_];

    const __bf16* ahp = h_hi + src + (size_t)(mh * 16 + n) * H_;  // A row = b
    const __bf16* alp = h_lo + src + (size_t)(mh * 16 + n) * H_;
#pragma unroll 4
    for (int kt = 0; kt < 32; ++kt) {
      const int k0 = kt * 32;
      v16bf ahi = loadA(ahp + k0, half);
      v16bf alo = loadA(alp + k0, half);
      v16bf bhi = loadB(bhp + k0, half);
      v16bf blo = loadB(blp + k0, half);
      acc = wmma_bf16(ahi, bhi, acc);
      acc = wmma_bf16(ahi, blo, acc);
      acc = wmma_bf16(alo, bhi, acc);
    }

    // stash this wave's gate tile; D rows b = mh*16 + half*8 + i
#pragma unroll
    for (int i = 0; i < 8; ++i) gates[g][mh * 16 + half * 8 + i][n] = acc[i];
    __syncthreads();

    // elementwise LSTM cell update; emit h directly as split bf16
    const size_t dst = (size_t)((t + 1) & 1) * (B_ * H_);
#pragma unroll
    for (int q = 0; q < 2; ++q) {
      const int idx = tid * 2 + q;       // 0..511
      const int b = idx >> 4, nn = idx & 15;
      float iv = gates[0][b][nn], fv = gates[1][b][nn];
      float gv = gates[2][b][nn], ov = gates[3][b][nn];
      iv = 1.f / (1.f + __expf(-iv));
      fv = 1.f / (1.f + __expf(-fv));
      gv = tanhf(gv);
      ov = 1.f / (1.f + __expf(-ov));
      float c = fv * cst[b][nn] + iv * gv;
      cst[b][nn] = c;
      float h = ov * tanhf(c);
      __bf16 hh = (__bf16)h;
      h_hi[dst + b * H_ + jb + nn] = hh;
      h_lo[dst + b * H_ + jb + nn] = (__bf16)(h - (float)hh);
      if (t == T_ - 1) out[b * H_ + jb + nn] = h;
    }

    // device-wide barrier: release h(t+1), wait for all 64 blocks
    __threadfence();
    __syncthreads();
    if (tid == 0) {
      unsigned gen = __hip_atomic_load(&bar[1], __ATOMIC_RELAXED,
                                       __HIP_MEMORY_SCOPE_AGENT);
      unsigned a = __hip_atomic_fetch_add(&bar[0], 1u, __ATOMIC_ACQ_REL,
                                          __HIP_MEMORY_SCOPE_AGENT);
      if (a == NBLK - 1) {
        __hip_atomic_store(&bar[0], 0u, __ATOMIC_RELAXED,
                           __HIP_MEMORY_SCOPE_AGENT);
        __hip_atomic_fetch_add(&bar[1], 1u, __ATOMIC_RELEASE,
                               __HIP_MEMORY_SCOPE_AGENT);
      } else {
        while (__hip_atomic_load(&bar[1], __ATOMIC_ACQUIRE,
                                 __HIP_MEMORY_SCOPE_AGENT) == gen) {
          __builtin_amdgcn_s_sleep(1);
        }
      }
    }
    __syncthreads();
    __threadfence();
  }
}

// ---------------------------------------------------------------------------
extern "C" void kernel_launch(void* const* d_in, const int* in_sizes, int n_in,
                              void* d_out, int out_size, void* d_ws, size_t ws_size,
                              hipStream_t stream) {
  const float* x    = (const float*)d_in[0];
  const float* w_ih = (const float*)d_in[1];
  const float* w_hh = (const float*)d_in[2];
  const float* b_ih = (const float*)d_in[3];
  const float* b_hh = (const float*)d_in[4];
  float* out = (float*)d_out;

  char* ws = (char*)d_ws;
  size_t off = 0;
  float* xg = (float*)(ws + off);            off += (size_t)T_ * B_ * G4H_ * 4;  // 256 MiB
  __bf16* x_hi = (__bf16*)(ws + off);        off += (size_t)B_ * T_ * H_ * 2;    // 32 MiB
  __bf16* x_lo = (__bf16*)(ws + off);        off += (size_t)B_ * T_ * H_ * 2;
  __bf16* wih_hi = (__bf16*)(ws + off);      off += (size_t)G4H_ * H_ * 2;       // 8 MiB
  __bf16* wih_lo = (__bf16*)(ws + off);      off += (size_t)G4H_ * H_ * 2;
  __bf16* whh_hi = (__bf16*)(ws + off);      off += (size_t)G4H_ * H_ * 2;
  __bf16* whh_lo = (__bf16*)(ws + off);      off += (size_t)G4H_ * H_ * 2;
  __bf16* h_hi = (__bf16*)(ws + off);        off += (size_t)2 * B_ * H_ * 2;     // 128 KiB
  __bf16* h_lo = (__bf16*)(ws + off);        off += (size_t)2 * B_ * H_ * 2;
  unsigned* bar = (unsigned*)(ws + off);

  // zero h buffers (hi starts right after whh_lo; hi+lo are contiguous)
  const int hwords = (int)((size_t)4 * B_ * H_ * 2 / 4);
  lstm_init<<<64, 256, 0, stream>>>((unsigned*)h_hi, hwords, bar);

  // one-time fp32 -> (hi,lo) bf16 splits
  split_f32_bf16<<<2048, 256, 0, stream>>>(x, x_hi, x_lo, B_ * T_ * H_);
  split_f32_bf16<<<1024, 256, 0, stream>>>(w_ih, wih_hi, wih_lo, G4H_ * H_);
  split_f32_bf16<<<1024, 256, 0, stream>>>(w_hh, whh_hi, whh_lo, G4H_ * H_);

  dim3 gridB(1024, 32);   // 1024 row tiles x (32*8) column tiles
  lstm_xg_gemm<<<gridB, 256, 0, stream>>>(x_hi, x_lo, wih_hi, wih_lo,
                                          b_ih, b_hh, xg);
  lstm_scan<<<NBLK, 256, 0, stream>>>(xg, whh_hi, whh_lo, h_hi, h_lo, bar, out);
}
